// LoRAMHSA_89489938579616
// MI455X (gfx1250) — matrix operand order repaired and checked by reading, not compiled, not run on hardware
//
#include <hip/hip_runtime.h>
#include <hip/hip_bf16.h>

typedef __bf16 v16bf __attribute__((ext_vector_type(16)));
typedef __bf16 bf16x8 __attribute__((ext_vector_type(8)));
typedef float  v8f   __attribute__((ext_vector_type(8)));
typedef int    v4i   __attribute__((vector_size(16)));

union V16 { v16bf v; bf16x8 h[2]; };

static __device__ __forceinline__ v8f wmma_bf16(const V16& a, const V16& b, v8f c) {
  return __builtin_amdgcn_wmma_f32_16x16x32_bf16(false, a.v, false, b.v, (short)0, c, false, false);
}

// ---- CDNA5 async global->LDS copy (ASYNCcnt) with synchronous fallback -------
#if defined(__AMDGCN__) && __has_builtin(__builtin_amdgcn_global_load_async_to_lds_b128)
#define HAVE_ASYNC_LDS 1
typedef __attribute__((address_space(1))) v4i* glb_v4i_p;
typedef __attribute__((address_space(3))) v4i* lds_v4i_p;
#define ASYNC_COPY16(ldsp, glbp)                                               \
  __builtin_amdgcn_global_load_async_to_lds_b128(                              \
      (glb_v4i_p)(glbp), (lds_v4i_p)(ldsp), 0, 0)
#if __has_builtin(__builtin_amdgcn_s_wait_asynccnt)
#define WAIT_ASYNC(n) __builtin_amdgcn_s_wait_asynccnt(n)
#else
#define WAIT_ASYNC(n) asm volatile("s_wait_asynccnt %0" ::"i"(n) : "memory")
#endif
#else
#define HAVE_ASYNC_LDS 0
#define ASYNC_COPY16(ldsp, glbp) (*(bf16x8*)(ldsp) = *(const bf16x8*)(glbp))
#define WAIT_ASYNC(n) ((void)0)
#endif

// ---------------------------------------------------------------- f32 -> bf16
__global__ __launch_bounds__(256) void cvt_f32_bf16(const float* __restrict__ src,
                                                    __bf16* __restrict__ dst, int n8) {
  int i = blockIdx.x * 256 + threadIdx.x;
  if (i >= n8) return;
  float4 a = ((const float4*)src)[2 * i];
  float4 b = ((const float4*)src)[2 * i + 1];
  bf16x8 o;
  o[0] = (__bf16)a.x; o[1] = (__bf16)a.y; o[2] = (__bf16)a.z; o[3] = (__bf16)a.w;
  o[4] = (__bf16)b.x; o[5] = (__bf16)b.y; o[6] = (__bf16)b.z; o[7] = (__bf16)b.w;
  ((bf16x8*)dst)[i] = o;
}

// ------------------------------------------------------- h = x . A[sid]^T  [M x 8]
template <typename TIN>
__global__ __launch_bounds__(256) void lora_h_kernel(const TIN* __restrict__ X,
                                                     const float* __restrict__ A,
                                                     const int* __restrict__ sid,
                                                     float* __restrict__ H,
                                                     int K, int rowsPerBatch) {
  int wid = threadIdx.x >> 5, lane = threadIdx.x & 31;
  int row = blockIdx.x * 8 + wid;
  int s = sid[row / rowsPerBatch];
  const TIN* xr = X + (size_t)row * K;
  const float* Ab = A + (size_t)s * 8 * K;
  float acc[8] = {0.f, 0.f, 0.f, 0.f, 0.f, 0.f, 0.f, 0.f};
  for (int d = lane; d < K; d += 32) {
    float xv = (float)xr[d];
#pragma unroll
    for (int r = 0; r < 8; ++r) acc[r] += xv * Ab[(size_t)r * K + d];
  }
#pragma unroll
  for (int r = 0; r < 8; ++r)
#pragma unroll
    for (int m = 16; m >= 1; m >>= 1) acc[r] += __shfl_xor(acc[r], m, 32);
  if (lane == 0) {
#pragma unroll
    for (int r = 0; r < 8; ++r) H[(size_t)row * 8 + r] = acc[r];
  }
}

// ------------------------- Out[MxN] = A[MxK](bf16) . W[NxK]^T + bias + 1/8 * h.Bl^T
// Double-buffered LDS tiles fed by async global->LDS DMA.
template <bool OUT_BF16>
__global__ __launch_bounds__(256) void gemm_lora_kernel(
    const __bf16* __restrict__ Ain, const __bf16* __restrict__ Wb,
    const float* __restrict__ bias, const float* __restrict__ hl,
    const float* __restrict__ Bl, const int* __restrict__ sid,
    void* __restrict__ Out, int M, int N, int K, int rowsPerBatch) {
  constexpr int BM = 128, BN = 128, BK = 32, LDT = 40;
  __shared__ __bf16 As[2][BM][LDT];
  __shared__ __bf16 Bs[2][BN][LDT];
  __shared__ float hS[BM * 8];
  __shared__ float bS[BN * 8];

  const int tid = threadIdx.x;
  const int lane = tid & 31, wid = tid >> 5;
  const int wm = wid & 3, wn = wid >> 2;        // 4 x 2 wave grid, 32x64 per wave
  const int half = lane >> 4, l15 = lane & 15;
  const int n0 = blockIdx.x * BN, m0 = blockIdx.y * BM;
  const int s = sid[m0 / rowsPerBatch];

  // stage rank-8 tables (contiguous 4KB each)
  ((float4*)hS)[tid] = ((const float4*)(hl + (size_t)m0 * 8))[tid];
  ((float4*)bS)[tid] = ((const float4*)(Bl + ((size_t)s * N + n0) * 8))[tid];

  // issue one A + one B 16-byte DMA per (thread, it); 4 async instrs per wave
  auto stage = [&](int buf, int kb) {
#pragma unroll
    for (int it = 0; it < 2; ++it) {
      int c = tid + it * 256;
      int r = c >> 2, c8 = (c & 3) * 8;
      ASYNC_COPY16(&As[buf][r][c8], Ain + (size_t)(m0 + r) * K + kb + c8);
      ASYNC_COPY16(&Bs[buf][r][c8], Wb + (size_t)(n0 + r) * K + kb + c8);
    }
  };

  v8f zero = {0.f, 0.f, 0.f, 0.f, 0.f, 0.f, 0.f, 0.f};
  v8f acc[2][4];
#pragma unroll
  for (int a = 0; a < 2; ++a)
#pragma unroll
    for (int b = 0; b < 4; ++b) acc[a][b] = zero;

  const int nk = K / BK;
  stage(0, 0);
  int cur = 0;
  for (int t = 0; t < nk; ++t) {
    if (t + 1 < nk) stage(cur ^ 1, (t + 1) * BK);  // prefetch next tile
    if (t + 1 < nk) { WAIT_ASYNC(4); } else { WAIT_ASYNC(0); }  // current tile landed
    __syncthreads();

    V16 af[2];
#pragma unroll
    for (int a = 0; a < 2; ++a) {
      int row = wm * 32 + a * 16 + l15;
      af[a].h[0] = *(const bf16x8*)&As[cur][row][half * 8];
      af[a].h[1] = *(const bf16x8*)&As[cur][row][16 + half * 8];
    }
    V16 bf_[4];
#pragma unroll
    for (int b = 0; b < 4; ++b) {
      int rn = wn * 64 + b * 16 + l15;
      bf_[b].h[0] = *(const bf16x8*)&Bs[cur][rn][half * 8];
      bf_[b].h[1] = *(const bf16x8*)&Bs[cur][rn][16 + half * 8];
    }
#pragma unroll
    for (int a = 0; a < 2; ++a)
#pragma unroll
      for (int b = 0; b < 4; ++b) acc[a][b] = wmma_bf16(af[a], bf_[b], acc[a][b]);
    __syncthreads();  // all waves done reading buf[cur] before it is re-filled
    cur ^= 1;
  }

  // epilogue: + bias + (ALPHA/RANK=0.125) * h . Bl row-dot
#pragma unroll
  for (int a = 0; a < 2; ++a)
#pragma unroll
    for (int b = 0; b < 4; ++b) {
      int colL = wn * 64 + b * 16 + l15;
      int col = n0 + colL;
      float bv = bias[col];
      const float* br = &bS[colL * 8];
#pragma unroll
      for (int j = 0; j < 8; ++j) {
        int rowL = wm * 32 + a * 16 + half * 8 + j;
        const float* hr = &hS[rowL * 8];
        float lora = 0.f;
#pragma unroll
        for (int r = 0; r < 8; ++r) lora += hr[r] * br[r];
        float v = acc[a][b][j] + bv + 0.125f * lora;
        size_t off = (size_t)(m0 + rowL) * N + col;
        if (OUT_BF16) ((__bf16*)Out)[off] = (__bf16)v;
        else          ((float*)Out)[off] = v;
      }
    }
}

// --------------------------------------------------- flash attention, bf16 WMMA
__global__ __launch_bounds__(256) void attn_kernel(const __bf16* __restrict__ qkv,
                                                   __bf16* __restrict__ y) {
  constexpr int T = 1024, C = 1024, C3 = 3072, DH = 64, BQ = 128, BKV = 64;
  __shared__ __bf16 Ks[BKV][72];
  __shared__ __bf16 Vt[DH][72];      // transposed: Vt[d][key]
  __shared__ __bf16 Ps[8][16][72];   // per-wave P tile

  const int tid = threadIdx.x, lane = tid & 31, wid = tid >> 5;
  const int half = lane >> 4, l15 = lane & 15;
  const int qblk = blockIdx.x & 7, h = (blockIdx.x >> 3) & 15, b = blockIdx.x >> 7;
  const int q0 = qblk * BQ, qr0 = q0 + wid * 16;

  const __bf16* qb = qkv + (size_t)b * T * C3 + h * DH;
  const __bf16* kb_ = qb + C;
  const __bf16* vb_ = qb + 2 * C;

  // q fragments (held in registers for all key blocks)
  V16 qf[2];
  {
    const __bf16* qrow = qb + (size_t)(qr0 + l15) * C3;
#pragma unroll
    for (int st = 0; st < 2; ++st) {
      qf[st].h[0] = *(const bf16x8*)(qrow + st * 32 + half * 8);
      qf[st].h[1] = *(const bf16x8*)(qrow + st * 32 + 16 + half * 8);
    }
  }

  v8f zero = {0.f, 0.f, 0.f, 0.f, 0.f, 0.f, 0.f, 0.f};
  v8f od[4];
#pragma unroll
  for (int dt = 0; dt < 4; ++dt) od[dt] = zero;
  float mi[8], li[8];
#pragma unroll
  for (int j = 0; j < 8; ++j) { mi[j] = -1e30f; li[j] = 0.f; }

  for (int kb = 0; kb < T; kb += BKV) {
    const __bf16* kbase = kb_ + (size_t)kb * C3;
    const __bf16* vbase = vb_ + (size_t)kb * C3;
#pragma unroll
    for (int it = 0; it < 2; ++it) {
      int c = tid + it * 256;
      int r = c >> 3, c8 = (c & 7) * 8;
      ASYNC_COPY16(&Ks[r][c8], kbase + (size_t)r * C3 + c8);  // K tile via DMA
      bf16x8 vv = *(const bf16x8*)(vbase + (size_t)r * C3 + c8);
#pragma unroll
      for (int j = 0; j < 8; ++j) Vt[c8 + j][r] = vv[j];      // V transposed
    }
    WAIT_ASYNC(0);
    __syncthreads();

    // S = q . k^T  (16 x 64 per wave)
    v8f s4[4];
#pragma unroll
    for (int nt = 0; nt < 4; ++nt) s4[nt] = zero;
#pragma unroll
    for (int st = 0; st < 2; ++st)
#pragma unroll
      for (int nt = 0; nt < 4; ++nt) {
        V16 bfr;
        int rn = nt * 16 + l15;
        bfr.h[0] = *(const bf16x8*)&Ks[rn][st * 32 + half * 8];
        bfr.h[1] = *(const bf16x8*)&Ks[rn][st * 32 + 16 + half * 8];
        s4[nt] = wmma_bf16(qf[st], bfr, s4[nt]);
      }

    // online softmax (rows j -> M = j + 8*half; cols = l15 within half)
    float esc[8];
#pragma unroll
    for (int j = 0; j < 8; ++j) {
      float mx = -1e30f;
#pragma unroll
      for (int nt = 0; nt < 4; ++nt) {
        s4[nt][j] *= 0.125f;               // 1/sqrt(64)
        mx = fmaxf(mx, s4[nt][j]);
      }
#pragma unroll
      for (int m = 8; m >= 1; m >>= 1) mx = fmaxf(mx, __shfl_xor(mx, m, 32));
      float mn = fmaxf(mi[j], mx);
      float e = __expf(mi[j] - mn);
      float ps = 0.f;
#pragma unroll
      for (int nt = 0; nt < 4; ++nt) {
        float p = __expf(s4[nt][j] - mn);
        s4[nt][j] = p;
        ps += p;
      }
#pragma unroll
      for (int m = 8; m >= 1; m >>= 1) ps += __shfl_xor(ps, m, 32);
      li[j] = li[j] * e + ps;
      mi[j] = mn;
      esc[j] = e;
    }
#pragma unroll
    for (int dt = 0; dt < 4; ++dt)
#pragma unroll
      for (int j = 0; j < 8; ++j) od[dt][j] *= esc[j];

    // restage P through LDS to get the A-fragment layout
#pragma unroll
    for (int nt = 0; nt < 4; ++nt)
#pragma unroll
      for (int j = 0; j < 8; ++j)
        Ps[wid][j + 8 * half][nt * 16 + l15] = (__bf16)s4[nt][j];

    // O += P . V
#pragma unroll
    for (int st = 0; st < 2; ++st) {
      V16 pa;
      pa.h[0] = *(const bf16x8*)&Ps[wid][l15][st * 32 + half * 8];
      pa.h[1] = *(const bf16x8*)&Ps[wid][l15][st * 32 + half * 8 + 16];
#pragma unroll
      for (int dt = 0; dt < 4; ++dt) {
        V16 vbr;
        int rd = dt * 16 + l15;
        vbr.h[0] = *(const bf16x8*)&Vt[rd][st * 32 + half * 8];
        vbr.h[1] = *(const bf16x8*)&Vt[rd][st * 32 + 16 + half * 8];
        od[dt] = wmma_bf16(pa, vbr, od[dt]);
      }
    }
    __syncthreads();
  }

  // finalize and scatter (bf16 output, [B,T,C] with head offset)
#pragma unroll
  for (int dt = 0; dt < 4; ++dt)
#pragma unroll
    for (int j = 0; j < 8; ++j) {
      float val = od[dt][j] / li[j];
      int row = qr0 + j + 8 * half;
      int d = dt * 16 + l15;
      y[((size_t)b * T + row) * C + h * DH + d] = (__bf16)val;
    }
}

// ------------------------------------------------------------------ launcher
extern "C" void kernel_launch(void* const* d_in, const int* in_sizes, int n_in,
                              void* d_out, int out_size, void* d_ws, size_t ws_size,
                              hipStream_t stream) {
  constexpr int B = 8, T = 1024, C = 1024, C3 = 3072, M = B * T;
  const float* x    = (const float*)d_in[0];
  // d_in[1] = pos_embedding (unused by the reference)
  const int*   sid  = (const int*)d_in[2];
  const float* Wqkv = (const float*)d_in[3];
  const float* bqkv = (const float*)d_in[4];
  const float* A1   = (const float*)d_in[5];
  const float* B1   = (const float*)d_in[6];
  const float* Wp   = (const float*)d_in[7];
  const float* bp   = (const float*)d_in[8];
  const float* A2   = (const float*)d_in[9];
  const float* B2   = (const float*)d_in[10];

  char* ws = (char*)d_ws;
  size_t off = 0;
  auto carve = [&](size_t bytes) { char* p = ws + off; off = (off + bytes + 255) & ~(size_t)255; return p; };
  __bf16* xb    = (__bf16*)carve((size_t)M * C * 2);
  __bf16* wqkvb = (__bf16*)carve((size_t)C3 * C * 2);
  __bf16* wpb   = (__bf16*)carve((size_t)C * C * 2);
  __bf16* qkvb  = (__bf16*)carve((size_t)M * C3 * 2);
  float*  h1    = (float*)carve((size_t)M * 8 * 4);
  __bf16* yat   = (__bf16*)carve((size_t)M * C * 2);
  float*  h2    = (float*)carve((size_t)M * 8 * 4);
  (void)ws_size; (void)n_in; (void)in_sizes; (void)out_size;

  auto cvt = [&](const float* s, __bf16* d, int n) {
    int n8 = n / 8;
    cvt_f32_bf16<<<(n8 + 255) / 256, 256, 0, stream>>>(s, d, n8);
  };
  cvt(x, xb, M * C);
  cvt(Wqkv, wqkvb, C3 * C);
  cvt(Wp, wpb, C * C);

  lora_h_kernel<float><<<M / 8, 256, 0, stream>>>(x, A1, sid, h1, C, T);
  gemm_lora_kernel<true><<<dim3(C3 / 128, M / 128), 256, 0, stream>>>(
      xb, wqkvb, bqkv, h1, B1, sid, qkvb, M, C3, C, T);

  attn_kernel<<<B * 16 * (T / 128), 256, 0, stream>>>(qkvb, yat);

  lora_h_kernel<__bf16><<<M / 8, 256, 0, stream>>>(yat, A2, sid, h2, C, T);
  gemm_lora_kernel<false><<<dim3(C / 128, M / 128), 256, 0, stream>>>(
      yat, wpb, bp, h2, B2, sid, d_out, M, C, C, T);
}
